// UrbanRadianceField_55465207661115
// MI455X (gfx1250) — matrix operand order repaired and checked by reading, not compiled, not run on hardware
//
#include <hip/hip_runtime.h>
#include <math.h>

// ---------------------------------------------------------------------------
// UrbanRadianceField on MI455X (gfx1250, wave32, WMMA)
// Fused radiance MLP (32768 points x 64 voxels) with v_wmma_f32_16x16x32_f16.
// B fragments: global_load_b128 from L2 (addrspace(1) views after anti-LICM
// laundering), batch-loaded 8 fragments ahead of the 8 WMMAs per k-chunk so
// loads pipeline (descending s_wait_loadcnt) instead of wait-0 per WMMA.
// ---------------------------------------------------------------------------

typedef __attribute__((ext_vector_type(16))) _Float16 v16h;
typedef __attribute__((ext_vector_type(8)))  float    v8f;

// global-address-space views (force global_load selection after ptr laundering)
typedef __attribute__((address_space(1))) const _Float16 gcf16;
typedef __attribute__((address_space(1))) const float    gcf32;
typedef __attribute__((address_space(1))) const v16h     gv16h;

#define V_    64
#define LD_   32
#define H_    128
#define R_    8
#define NV_   512
#define VF_   64
#define NF_   10
#define VE_   60
#define B_    512
#define S_    64
#define N_    (B_ * S_)
#define NEAR_ 0.1f
#define FAR_  100.0f

// A-fragment swizzle for 16-bit 16x32 A tiles (ISA 7.12.2):
//  lane 0-15 holds row M, K {0..7,16..23}; lane 16-31 holds K {8..15,24..31}.
// Buffer layout [kchunk][lane(32)][slot(16)] -> per-lane fragment contiguous 32B.
__device__ __host__ __forceinline__ int aswz(int mloc, int k) {
  int kc   = k >> 5;
  int kl   = k & 31;
  int hi   = (kl >> 3) & 1;
  int slot = (kl & 7) | (((kl >> 4) & 1) << 3);
  int lane = mloc | (hi << 4);
  return kc * 512 + lane * 16 + slot;
}

// B-fragment swizzle for 16-bit 32x16 B tiles:
//  lane 0-15 = col n with K {0..7,16..23} (pairs per vgpr); lane 16-31 = K {8..15,24..31}
__global__ void k_swzB(const float* __restrict__ src, _Float16* __restrict__ dst,
                       int K, int N, int Ksrc, int Nsrc, int KC) {
  int idx = blockIdx.x * blockDim.x + threadIdx.x;
  if (idx >= K * N) return;
  int k = idx / N, n = idx - k * N;
  float val = (k < Ksrc && n < Nsrc) ? src[k * Nsrc + n] : 0.0f;
  int nt = n >> 4, c = n & 15, kc = k >> 5, kl = k & 31;
  int hi   = (kl >> 3) & 1;
  int slot = ((kl & 7) >> 1) | (((kl >> 4) & 1) << 2);
  int sub  = kl & 1;
  int lane = c | (hi << 4);
  dst[(((nt * KC + kc) * 32 + lane) * 16) + slot * 2 + sub] = (_Float16)val;
}

// Head weights: rgb(3) | density(1) | pad -> one 128x16 B panel, KC = 4
__global__ void k_swzW4(const float* __restrict__ wrgb, const float* __restrict__ wd,
                        _Float16* __restrict__ dst) {
  int idx = blockIdx.x * blockDim.x + threadIdx.x;
  if (idx >= 128 * 16) return;
  int k = idx >> 4, n = idx & 15;
  float val = (n < 3) ? wrgb[k * 3 + n] : ((n == 3) ? wd[k] : 0.0f);
  int kc = k >> 5, kl = k & 31;
  int hi   = (kl >> 3) & 1;
  int slot = ((kl & 7) >> 1) | (((kl >> 4) & 1) << 2);
  int sub  = kl & 1;
  int lane = n | (hi << 4);
  dst[((kc * 32 + lane) * 16) + slot * 2 + sub] = (_Float16)val;
}

// ---------------------------------------------------------------------------
// Per-voxel deformation MLP (batch-1 matvec, VALU f32) -> verts + |v|^2
// ---------------------------------------------------------------------------
__global__ __launch_bounds__(128)
void k_deform(const float* __restrict__ latents,
              const float* __restrict__ dW1, const float* __restrict__ db1,
              const float* __restrict__ dW2, const float* __restrict__ db2,
              const float* __restrict__ dW3, const float* __restrict__ db3,
              const float* __restrict__ dW4, const float* __restrict__ db4,
              float4* __restrict__ verts4) {
  __shared__ float sl[LD_];
  __shared__ float sa[H_];
  __shared__ float sb[H_];
  __shared__ float sv[NV_ * 3];
  const int v = blockIdx.x;
  const int t = threadIdx.x;
  if (t < LD_) sl[t] = latents[v * LD_ + t];
  __syncthreads();
  {
    float a = db1[v * H_ + t];
    for (int i = 0; i < LD_; ++i) a += sl[i] * dW1[(size_t)v * LD_ * H_ + i * H_ + t];
    sa[t] = fmaxf(a, 0.0f);
  }
  __syncthreads();
  {
    float a = db2[v * H_ + t];
    for (int i = 0; i < H_; ++i) a += sa[i] * dW2[(size_t)v * H_ * H_ + i * H_ + t];
    sb[t] = fmaxf(a, 0.0f);
  }
  __syncthreads();
  {
    float a = db3[v * H_ + t];
    for (int i = 0; i < H_; ++i) a += sb[i] * dW3[(size_t)v * H_ * H_ + i * H_ + t];
    sa[t] = fmaxf(a, 0.0f);
  }
  __syncthreads();
  for (int o = t; o < NV_ * 3; o += 128) {
    float a = db4[(size_t)v * NV_ * 3 + o];
    for (int k = 0; k < H_; ++k)
      a += sa[k] * dW4[(size_t)v * H_ * NV_ * 3 + (size_t)k * NV_ * 3 + o];
    int vid = o / 3, c = o - vid * 3;
    int ig = (c == 0) ? (vid >> 6) : (c == 1 ? ((vid >> 3) & 7) : (vid & 7));
    float base = -1.0f + (2.0f / (float)(R_ - 1)) * (float)ig;
    sv[o] = base + a;
  }
  __syncthreads();
  for (int p = t; p < NV_; p += 128) {
    float x = sv[3 * p], y = sv[3 * p + 1], z = sv[3 * p + 2];
    verts4[(size_t)v * NV_ + p] = make_float4(x, y, z, x * x + y * y + z * z);
  }
}

// ---------------------------------------------------------------------------
// Positional encoding of view directions: [B, 64] (60 used + pad)
// ---------------------------------------------------------------------------
__global__ void k_view(const float* __restrict__ rd, float* __restrict__ viewenc) {
  int r = blockIdx.x * blockDim.x + threadIdx.x;
  if (r >= B_) return;
  float d3[3] = {rd[r * 3], rd[r * 3 + 1], rd[r * 3 + 2]};
  for (int f = 0; f < NF_; ++f) {
    float fs = (float)(1 << f);
    for (int c = 0; c < 3; ++c) {
      float xe = d3[c] * fs;
      viewenc[r * 64 + f * 6 + c]     = sinf(xe);
      viewenc[r * 64 + f * 6 + 3 + c] = cosf(xe);
    }
  }
  for (int c = 60; c < 64; ++c) viewenc[r * 64 + c] = 0.0f;
}

// ---------------------------------------------------------------------------
// One 8-tile WMMA accumulation pass. A chunks kc<4 from aLo, kc>=4 from aHi.
// All 8 B fragments of a k-chunk are loaded into distinct values BEFORE the
// 8 WMMAs consume them -> backend keeps 16 global_load_b128 in flight and
// issues descending s_wait_loadcnt instead of wait-0 per WMMA.
// ---------------------------------------------------------------------------
template <int KC>
__device__ __forceinline__ void wmma_pass8(const _Float16* aLo, const _Float16* aHi,
                                           gcf16* bz,
                                           int ntBase, int lane, v8f* acc) {
#pragma unroll
  for (int kc = 0; kc < KC; ++kc) {
    const _Float16* ap = (kc < 4) ? (aLo + kc * 512) : (aHi + (kc - 4) * 512);
    v16h A = *(const v16h*)(ap + lane * 16);
    v16h Bf[8];
#pragma unroll
    for (int j = 0; j < 8; ++j)
      Bf[j] = *(gv16h*)(bz + ((size_t)((ntBase + j) * KC + kc) * 32 + lane) * 16);
#pragma unroll
    for (int j = 0; j < 8; ++j)
      acc[j] = __builtin_amdgcn_wmma_f32_16x16x32_f16(
          false, A, false, Bf[j], (short)0, acc[j], false, false);
  }
}

// bias + relu + f16 convert + A-swizzled store (k index = n - kOffset)
__device__ __forceinline__ void epilogue8(const v8f* acc, int ntBase,
                                          gcf32* bias,
                                          _Float16* dst, int kOffset, int lane) {
  const int n4  = lane & 15;
  const int rhi = (lane >> 4) << 3;
#pragma unroll
  for (int j = 0; j < 8; ++j) {
    const int n  = (ntBase + j) * 16 + n4;
    const float bb = bias[n];
#pragma unroll
    for (int r = 0; r < 8; ++r) {
      float val = fmaxf(acc[j][r] + bb, 0.0f);
      dst[aswz(r + rhi, n - kOffset)] = (_Float16)val;
    }
  }
}

// ---------------------------------------------------------------------------
// Main fused kernel: argmin + gather + 4 GEMMs per (64-point tile, voxel)
// ---------------------------------------------------------------------------
__global__ __launch_bounds__(128)
void k_main(const float* __restrict__ ro, const float* __restrict__ rd,
            const float* __restrict__ vfeats,
            const float4* __restrict__ verts4,
            const float* __restrict__ viewenc,
            const _Float16* __restrict__ wz1, const _Float16* __restrict__ wz2,
            const _Float16* __restrict__ wz3, const _Float16* __restrict__ wz4,
            const float* __restrict__ rb1, const float* __restrict__ rb2,
            const float* __restrict__ rb3, const float* __restrict__ rbrgb,
            const float* __restrict__ rbd,
            float* __restrict__ rgb_sum, float* __restrict__ den_sum) {
  __shared__ float4 sh_verts[NV_];
  __shared__ float  sh_bd[128];
  __shared__ int    sh_bi[128];
  __shared__ int    sh_idx[64];
  __shared__ __align__(32) _Float16 sh_act[4 * 4096];  // main: 8KB / wave
  __shared__ __align__(32) _Float16 sh_aux[4 * 2048];  // aux:  4KB / wave

  const int tid  = threadIdx.x;
  const int lane = tid & 31;
  const int wave = tid >> 5;
  const int m0   = blockIdx.x * 64;
  const int n4   = lane & 15;
  const int rhi  = (lane >> 4) << 3;

  // argmin geometry: 2 threads per point (each scans 256 verts)
  const int pl    = tid & 63;
  const int vhalf = tid >> 6;
  const int mArg  = m0 + pl;
  const int bArg  = mArg / S_;
  const float tS  = NEAR_ + (FAR_ - NEAR_) * (float)(mArg % S_) * (1.0f / (S_ - 1));
  const float px  = ro[bArg * 3 + 0] + rd[bArg * 3 + 0] * tS;
  const float py  = ro[bArg * 3 + 1] + rd[bArg * 3 + 1] * tS;
  const float pz  = ro[bArg * 3 + 2] + rd[bArg * 3 + 2] * tS;

  gcf32* vfg = (gcf32*)(const float*)vfeats;

  _Float16* bufw = &sh_act[wave * 4096];
  _Float16* auxw = &sh_aux[wave * 2048];
  float sum[8] = {0.f, 0.f, 0.f, 0.f, 0.f, 0.f, 0.f, 0.f};
  const v8f z8 = {0.f, 0.f, 0.f, 0.f, 0.f, 0.f, 0.f, 0.f};

  for (int v = 0; v < V_; ++v) {
    // ---- anti-LICM barrier: weight/bias bases are "redefined" each voxel so
    // the ~260 B-fragment global loads are NOT hoisted out of the loop (which
    // would spill them all to scratch). Emits zero instructions. The explicit
    // addrspace(1) casts below restore global_load selection.
    unsigned long long a1 = (unsigned long long)wz1;
    unsigned long long a2 = (unsigned long long)wz2;
    unsigned long long a3 = (unsigned long long)wz3;
    unsigned long long a4 = (unsigned long long)wz4;
    unsigned long long b1 = (unsigned long long)rb1;
    unsigned long long b2 = (unsigned long long)rb2;
    unsigned long long b3 = (unsigned long long)rb3;
    asm volatile("" : "+s"(a1), "+s"(a2), "+s"(a3), "+s"(a4),
                      "+s"(b1), "+s"(b2), "+s"(b3));
    gcf16* wz1v = (gcf16*)(const _Float16*)a1;
    gcf16* wz2v = (gcf16*)(const _Float16*)a2;
    gcf16* wz3v = (gcf16*)(const _Float16*)a3;
    gcf16* wz4v = (gcf16*)(const _Float16*)a4;
    gcf32* rb1v = (gcf32*)(const float*)b1;
    gcf32* rb2v = (gcf32*)(const float*)b2;
    gcf32* rb3v = (gcf32*)(const float*)b3;

    __syncthreads();  // protect sh_verts/sh_act from previous iteration
    for (int i = tid; i < NV_; i += 128) sh_verts[i] = verts4[(size_t)v * NV_ + i];
    if (v + 1 < V_) {
      __builtin_prefetch(&verts4[(size_t)(v + 1) * NV_], 0, 0);
      __builtin_prefetch(&vfeats[(size_t)(v + 1) * NV_ * VF_], 0, 0);
    }
    __syncthreads();

    // --- nearest vertex: argmin_j (|v_j|^2 - 2 p.v_j); |p|^2 constant ---
    float bd = 3.4e38f;
    int   bi = 0;
    const int j0 = vhalf * 256;
    for (int j = j0; j < j0 + 256; ++j) {
      float4 w  = sh_verts[j];
      float  dd = w.w - 2.0f * (px * w.x + py * w.y + pz * w.z);
      if (dd < bd) { bd = dd; bi = j; }
    }
    sh_bd[tid] = bd; sh_bi[tid] = bi;
    __syncthreads();
    if (tid < 64) {
      float d0 = sh_bd[tid], d1 = sh_bd[tid + 64];
      sh_idx[tid] = (d1 < d0) ? sh_bi[tid + 64] : sh_bi[tid];
    }
    __syncthreads();

    // --- build x = [feats | viewenc | 0] into A-swizzled LDS, column = tid ---
    {
      const int k = tid;
      const unsigned vbase = (unsigned)v * (NV_ * VF_);
      for (int row = 0; row < 64; ++row) {
        float val;
        if (k < VF_)
          val = vfg[vbase + (unsigned)sh_idx[row] * VF_ + (unsigned)k];
        else if (k < VF_ + VE_)
          val = viewenc[((m0 + row) / S_) * 64 + (k - VF_)];
        else
          val = 0.0f;
        sh_act[(row >> 4) * 4096 + aswz(row & 15, k)] = (_Float16)val;
      }
    }
    __syncthreads();

    v8f acc[8];

    // --- GEMM1: x[16x128] x W1[128x256] -> h1 (main kc0..7) ---
    // tiles 8..15 first: writes kc4..7 (disjoint from x in kc0..3)
#pragma unroll
    for (int i = 0; i < 8; ++i) acc[i] = z8;
    wmma_pass8<4>(bufw, bufw + 2048, wz1v, 8, lane, acc);
    epilogue8(acc, 8, rb1v, bufw, 0, lane);
    // tiles 0..7: reads x (kc0..3, still intact), then overwrites kc0..3
#pragma unroll
    for (int i = 0; i < 8; ++i) acc[i] = z8;
    wmma_pass8<4>(bufw, bufw + 2048, wz1v, 0, lane, acc);
    epilogue8(acc, 0, rb1v, bufw, 0, lane);

    // --- GEMM2: h1[16x256] x W2[256x256] -> h2 (main kc0..3 + aux kc0..3) ---
    // tiles 8..15 -> aux (k index 128..255 remapped to aux kc0..3)
#pragma unroll
    for (int i = 0; i < 8; ++i) acc[i] = z8;
    wmma_pass8<8>(bufw, bufw + 2048, wz2v, 8, lane, acc);
    epilogue8(acc, 8, rb2v, auxw, 128, lane);
    // tiles 0..7 -> main kc0..3 (h1 fully read before writes)
#pragma unroll
    for (int i = 0; i < 8; ++i) acc[i] = z8;
    wmma_pass8<8>(bufw, bufw + 2048, wz2v, 0, lane, acc);
    epilogue8(acc, 0, rb2v, bufw, 0, lane);

    // --- GEMM3: h2[16x256] x W3[256x128] -> h3 (main kc0..3) ---
#pragma unroll
    for (int i = 0; i < 8; ++i) acc[i] = z8;
    wmma_pass8<8>(bufw, auxw, wz3v, 0, lane, acc);
    epilogue8(acc, 0, rb3v, bufw, 0, lane);

    // --- GEMM4: h3[16x128] x W4[128x16] -> rgb(0..2) | density(3) ---
    {
      v16h A4[4], B4[4];
#pragma unroll
      for (int kc = 0; kc < 4; ++kc) {
        A4[kc] = *(const v16h*)(bufw + kc * 512 + lane * 16);
        B4[kc] = *(gv16h*)(wz4v + ((size_t)(kc * 32 + lane)) * 16);
      }
      v8f a4v = z8;
#pragma unroll
      for (int kc = 0; kc < 4; ++kc)
        a4v = __builtin_amdgcn_wmma_f32_16x16x32_f16(
            false, A4[kc], false, B4[kc], (short)0, a4v, false, false);
      float bias = (n4 < 3) ? rbrgb[n4] : ((n4 == 3) ? rbd[0] : 0.0f);
#pragma unroll
      for (int r = 0; r < 8; ++r) {
        float val = a4v[r] + bias;
        val = (n4 == 3) ? fmaxf(val, 0.0f) : (1.0f / (1.0f + __expf(-val)));
        sum[r] += val;
      }
    }
  }

  // write per-point voxel-averaged rgb / density
#pragma unroll
  for (int r = 0; r < 8; ++r) {
    int mg = m0 + wave * 16 + r + rhi;
    float val = sum[r] * (1.0f / (float)V_);
    if (n4 < 3)       rgb_sum[mg * 3 + n4] = val;
    else if (n4 == 3) den_sum[mg] = val;
  }
}

// ---------------------------------------------------------------------------
// Volume rendering (sequential alpha compositing per ray)
// ---------------------------------------------------------------------------
__global__ void k_render(const float* __restrict__ rgb_sum,
                         const float* __restrict__ den_sum,
                         float* __restrict__ out) {
  int b = blockIdx.x * blockDim.x + threadIdx.x;
  if (b >= B_) return;
  float* rgbF = out;                 // [B,3]
  float* depF = out + B_ * 3;        // [B]
  float* wF   = out + B_ * 3 + B_;   // [B,S]
  const float dt = (FAR_ - NEAR_) / (float)(S_ - 1);
  float T = 1.0f, r = 0.f, g = 0.f, bl = 0.f, dep = 0.f;
  for (int s = 0; s < S_; ++s) {
    int m = b * S_ + s;
    float t     = NEAR_ + dt * (float)s;
    float delta = (s == S_ - 1) ? 1e10f : dt;
    float alpha = 1.0f - __expf(-den_sum[m] * delta);
    float w     = alpha * T;
    r  += w * rgb_sum[m * 3 + 0];
    g  += w * rgb_sum[m * 3 + 1];
    bl += w * rgb_sum[m * 3 + 2];
    dep += w * t;
    wF[b * S_ + s] = w;
    T *= (1.0f - alpha + 1e-10f);
  }
  rgbF[b * 3 + 0] = r;
  rgbF[b * 3 + 1] = g;
  rgbF[b * 3 + 2] = bl;
  depF[b] = dep;
}

// ---------------------------------------------------------------------------
extern "C" void kernel_launch(void* const* d_in, const int* in_sizes, int n_in,
                              void* d_out, int out_size, void* d_ws, size_t ws_size,
                              hipStream_t stream) {
  (void)in_sizes; (void)n_in; (void)out_size; (void)ws_size;
  const float* ro    = (const float*)d_in[0];
  const float* rd    = (const float*)d_in[1];
  const float* lat   = (const float*)d_in[2];
  const float* dW1   = (const float*)d_in[3];
  const float* db1   = (const float*)d_in[4];
  const float* dW2   = (const float*)d_in[5];
  const float* db2   = (const float*)d_in[6];
  const float* dW3   = (const float*)d_in[7];
  const float* db3   = (const float*)d_in[8];
  const float* dW4   = (const float*)d_in[9];
  const float* db4   = (const float*)d_in[10];
  const float* vfeat = (const float*)d_in[11];
  const float* rW1   = (const float*)d_in[12];
  const float* rb1   = (const float*)d_in[13];
  const float* rW2   = (const float*)d_in[14];
  const float* rb2   = (const float*)d_in[15];
  const float* rW3   = (const float*)d_in[16];
  const float* rb3   = (const float*)d_in[17];
  const float* rWrgb = (const float*)d_in[18];
  const float* rbrgb = (const float*)d_in[19];
  const float* rWd   = (const float*)d_in[20];
  const float* rbd   = (const float*)d_in[21];

  char* ws = (char*)d_ws;
  float4*    verts4  = (float4*)(ws + 0);           // 64*512*16  = 524288
  float*     viewenc = (float*)(ws + 524288);       // 512*64*4   = 131072
  _Float16*  wz1     = (_Float16*)(ws + 655360);    // 65536
  _Float16*  wz2     = (_Float16*)(ws + 720896);    // 131072
  _Float16*  wz3     = (_Float16*)(ws + 851968);    // 65536
  _Float16*  wz4     = (_Float16*)(ws + 917504);    // 4096
  float*     rgb_sum = (float*)(ws + 921600);       // 32768*3*4  = 393216
  float*     den_sum = (float*)(ws + 1314816);      // 32768*4    = 131072

  // weight reformat (f32 -> f16 B-fragments)
  k_swzB<<<(128 * 256 + 255) / 256, 256, 0, stream>>>(rW1, wz1, 128, 256, 124, 256, 4);
  k_swzB<<<(256 * 256 + 255) / 256, 256, 0, stream>>>(rW2, wz2, 256, 256, 256, 256, 8);
  k_swzB<<<(256 * 128 + 255) / 256, 256, 0, stream>>>(rW3, wz3, 256, 128, 256, 128, 8);
  k_swzW4<<<(128 * 16 + 255) / 256, 256, 0, stream>>>(rWrgb, rWd, wz4);

  // per-voxel deformation MLPs -> verts + norms
  k_deform<<<V_, 128, 0, stream>>>(lat, dW1, db1, dW2, db2, dW3, db3, dW4, db4, verts4);
  // view-direction positional encoding
  k_view<<<(B_ + 255) / 256, 256, 0, stream>>>(rd, viewenc);

  // fused argmin + gather + radiance MLP over all voxels
  k_main<<<N_ / 64, 128, 0, stream>>>(ro, rd, vfeat, verts4, viewenc,
                                      wz1, wz2, wz3, wz4,
                                      rb1, rb2, rb3, rbrgb, rbd,
                                      rgb_sum, den_sum);
  // volume rendering
  k_render<<<(B_ + 255) / 256, 256, 0, stream>>>(rgb_sum, den_sum, (float*)d_out);
}
